// CrossAttentionBlock_53008486367691
// MI455X (gfx1250) — compile-verified, matrix operand order checked by reading, not run on hardware
//
#include <hip/hip_runtime.h>
#include <hip/hip_bf16.h>

// CDNA5 / gfx1250: wave32, WMMA f32_16x16x32_f16 path.
// B=8, C=512, H=W=64 -> L=4096, 8 heads of d=64, channel-channel attention.
//
// Roofline: 73 GFLOP total; fp32 WMMA is 16x16x4 (8x fewer MACs/op) so all
// GEMMs run f16 WMMA with f32 accumulation. 320KB LDS/WGP lets each block keep
// the full K=512 x N=128 X^T panel resident (130KB f16): X is read/converted
// once, and only the small W tile is restaged per K-step (double-buffered,
// one barrier per step).

typedef __attribute__((ext_vector_type(16))) _Float16 v16h;
typedef __attribute__((ext_vector_type(8)))  _Float16 v8h;
typedef __attribute__((ext_vector_type(8)))  float    v8f;

#define BB 8
#define CC 512
#define LL 4096
#define NHEAD 8
#define DHEAD 64

__device__ __forceinline__ v8f zero8() {
  v8f z = {0.f,0.f,0.f,0.f,0.f,0.f,0.f,0.f};
  return z;
}

__device__ __forceinline__ v8f wmma16(v16h a, v16h b, v8f c) {
  // (neg_a, A, neg_b, B, c_mod, C, reuse_a, reuse_b) -> v_wmma_f32_16x16x32_f16
  return __builtin_amdgcn_wmma_f32_16x16x32_f16(false, a, false, b, (short)0, c,
                                                false, false);
}

__device__ __forceinline__ v16h mk16(v8h lo, v8h hi) {
  union { v16h v; v8h h[2]; } u;
  u.h[0] = lo; u.h[1] = hi;
  return u.v;
}

// WMMA 16x16x32 f16 A/B fragment loader from a row-major f16 matrix.
// Lane layout (ISA 7.12.2, 16-bit A 16x32): row = lane%16, h = lane/16,
// elements 0..7  -> K = kb + h*8 + i
// elements 8..15 -> K = kb + 16 + h*8 + (i-8)
// (B fragment is the mirrored layout with "row" = output column, so loading
//  the row-major transpose of B with this same helper is exactly right.)
__device__ __forceinline__ v16h frag_ld(const _Float16* base, int row0,
                                        int rowStride, int kb) {
  const int lane = threadIdx.x & 31;
  const _Float16* p = base + (size_t)(row0 + (lane & 15)) * rowStride
                           + kb + ((lane >> 4) << 3);
  v8h lo = *(const v8h*)(p);
  v8h hi = *(const v8h*)(p + 16);
  return mk16(lo, hi);
}

// ---------------------------------------------------------------------------
// Projection GEMM: Y[b,m,n] = W . X[b,:,n] + bias   (M=K=512, N=4096/batch)
// Block: 256 threads (8 waves), tile 128(M) x 128(N).
// The entire X^T panel (N=128 rows x K=512 cols, f16) is staged in LDS once;
// W tiles (128x32) are double-buffered per K-step so their global loads hide
// under the WMMAs with a single barrier per step.
// Wave (wm 0..3, wn 0..1) owns a 32x64 sub-tile = 2x4 WMMA accumulators.
// ---------------------------------------------------------------------------
template<bool IN_F16, bool OUT_F32>
__global__ __launch_bounds__(256) void panel_gemm(
    const float* __restrict__ W, const float* __restrict__ bias,
    const float* __restrict__ Xf, const _Float16* __restrict__ Xh,
    float* __restrict__ Yf, _Float16* __restrict__ Yh)
{
  __shared__ __align__(16) _Float16 sX[128][520];    // [n][k], 520 = 512+8 pad
  __shared__ __align__(16) _Float16 sA[2][128][40];  // double-buffered W tile
  __shared__ float sBias[128];

  const int tid  = threadIdx.x;
  const int b    = blockIdx.z;
  const int m0   = blockIdx.y * 128;
  const int n0   = blockIdx.x * 128;
  const int wid  = tid >> 5;
  const int lane = tid & 31;
  const int wm   = wid & 3;
  const int wn   = wid >> 2;

  if (tid < 128) sBias[tid] = bias[m0 + tid];

  // ---- stage X^T panel: X[b][k][n0..+127] -> sX[n][k] f16 ----
  // Each thread owns a K-pair (k2,k2+1) x 8 n's per 32-K chunk; the pair packs
  // into one 32-bit LDS store (halves the transpose scatter).
  {
    const int k2   = ((tid >> 4) & 15) * 2;  // 0,2,...,30
    const int nseg = tid & 15;               // n = nseg*8 .. +7
    for (int kk = 0; kk < 512; kk += 32) {
      if (IN_F16) {
        const _Float16* r0 =
            Xh + ((size_t)b * 512 + kk + k2) * LL + n0 + nseg * 8;
        v8h x0 = *(const v8h*)(r0);
        v8h x1 = *(const v8h*)(r0 + LL);
        if (kk + 32 < 512) __builtin_prefetch(r0 + (size_t)32 * LL, 0, 3);
        #pragma unroll
        for (int j = 0; j < 8; ++j) {
          union { _Float16 h[2]; unsigned u; } p;
          p.h[0] = x0[j];
          p.h[1] = x1[j];
          *(unsigned*)&sX[nseg * 8 + j][kk + k2] = p.u;  // k2 even: 4B aligned
        }
      } else {
        const float* r0 = Xf + ((size_t)b * 512 + kk + k2) * LL + n0 + nseg * 8;
        const float* r1 = r0 + LL;
        float4 a0 = *(const float4*)(r0);
        float4 a1 = *(const float4*)(r0 + 4);
        float4 c0 = *(const float4*)(r1);
        float4 c1 = *(const float4*)(r1 + 4);
        if (kk + 32 < 512) __builtin_prefetch(r0 + (size_t)32 * LL, 0, 3);
        float f0[8] = {a0.x,a0.y,a0.z,a0.w,a1.x,a1.y,a1.z,a1.w};
        float f1[8] = {c0.x,c0.y,c0.z,c0.w,c1.x,c1.y,c1.z,c1.w};
        #pragma unroll
        for (int j = 0; j < 8; ++j) {
          union { _Float16 h[2]; unsigned u; } p;
          p.h[0] = (_Float16)f0[j];
          p.h[1] = (_Float16)f1[j];
          *(unsigned*)&sX[nseg * 8 + j][kk + k2] = p.u;
        }
      }
    }
  }

  // ---- stage first W tile into sA[0] ----
  const int rA = tid >> 1, segA = tid & 1;
  {
    const float* src = W + (size_t)(m0 + rA) * 512 + segA * 16;
    float4 f0 = *(const float4*)(src + 0);
    float4 f1 = *(const float4*)(src + 4);
    float4 f2 = *(const float4*)(src + 8);
    float4 f3 = *(const float4*)(src + 12);
    v8h h0 = {(_Float16)f0.x,(_Float16)f0.y,(_Float16)f0.z,(_Float16)f0.w,
              (_Float16)f1.x,(_Float16)f1.y,(_Float16)f1.z,(_Float16)f1.w};
    v8h h1 = {(_Float16)f2.x,(_Float16)f2.y,(_Float16)f2.z,(_Float16)f2.w,
              (_Float16)f3.x,(_Float16)f3.y,(_Float16)f3.z,(_Float16)f3.w};
    *(v8h*)&sA[0][rA][segA * 16]     = h0;
    *(v8h*)&sA[0][rA][segA * 16 + 8] = h1;
  }
  __syncthreads();

  v8f acc[2][4];
  #pragma unroll
  for (int mt = 0; mt < 2; ++mt)
    #pragma unroll
    for (int nt = 0; nt < 4; ++nt) acc[mt][nt] = zero8();

  for (int ks = 0; ks < 16; ++ks) {
    const int kb  = ks * 32;
    const int cur = ks & 1;
    const bool have_next = (ks + 1 < 16);

    // issue global loads of the next W tile first (latency hides under WMMA)
    float4 f0 = {}, f1 = {}, f2 = {}, f3 = {};
    if (have_next) {
      const float* src = W + (size_t)(m0 + rA) * 512 + kb + 32 + segA * 16;
      f0 = *(const float4*)(src + 0);
      f1 = *(const float4*)(src + 4);
      f2 = *(const float4*)(src + 8);
      f3 = *(const float4*)(src + 12);
    }

    // compute: 2x4 WMMA tiles per wave from resident panel + current W tile
    v16h afr[2];
    #pragma unroll
    for (int mt = 0; mt < 2; ++mt)
      afr[mt] = frag_ld(&sA[cur][0][0], wm * 32 + mt * 16, 40, 0);
    #pragma unroll
    for (int nt = 0; nt < 4; ++nt) {
      v16h bfr = frag_ld(&sX[0][0], wn * 64 + nt * 16, 520, kb);
      #pragma unroll
      for (int mt = 0; mt < 2; ++mt)
        acc[mt][nt] = wmma16(afr[mt], bfr, acc[mt][nt]);
    }

    // convert + store next W tile into the other buffer
    if (have_next) {
      v8h h0 = {(_Float16)f0.x,(_Float16)f0.y,(_Float16)f0.z,(_Float16)f0.w,
                (_Float16)f1.x,(_Float16)f1.y,(_Float16)f1.z,(_Float16)f1.w};
      v8h h1 = {(_Float16)f2.x,(_Float16)f2.y,(_Float16)f2.z,(_Float16)f2.w,
                (_Float16)f3.x,(_Float16)f3.y,(_Float16)f3.z,(_Float16)f3.w};
      *(v8h*)&sA[cur ^ 1][rA][segA * 16]     = h0;
      *(v8h*)&sA[cur ^ 1][rA][segA * 16 + 8] = h1;
    }
    __syncthreads();
  }

  // ---- epilogue: + bias, store (C/D layout: VGPR r -> M=r(+8), N=lane%16) ----
  #pragma unroll
  for (int mt = 0; mt < 2; ++mt)
    #pragma unroll
    for (int r = 0; r < 8; ++r) {
      const int crel = wm * 32 + mt * 16 + r + ((lane >> 4) << 3);
      const float bv = sBias[crel];
      const int c = m0 + crel;
      #pragma unroll
      for (int nt = 0; nt < 4; ++nt) {
        const int n = n0 + wn * 64 + nt * 16 + (lane & 15);
        const float val = acc[mt][nt][r] + bv;
        const size_t off = ((size_t)b * 512 + c) * LL + n;
        if (OUT_F32) Yf[off] = val;
        else         Yh[off] = (_Float16)val;
      }
    }
}

// ---------------------------------------------------------------------------
// Channel attention, one block per (batch, head).
// Phase 1: S = q . k^T (64x64, K=4096) via WMMA, straight from global f16
//          (rows of q_head/k_head are K-contiguous, matching frag layout).
// Softmax over rows (scale 1/sqrt(L) = 1/64) in fp32 LDS -> P f16.
// Phase 2: O = P . v_head (64x4096, K=64), v tiles transposed through LDS.
// ---------------------------------------------------------------------------
__global__ __launch_bounds__(256) void attn64(
    const _Float16* __restrict__ q, const _Float16* __restrict__ k,
    const _Float16* __restrict__ v, _Float16* __restrict__ o)
{
  __shared__ float                  sS[64][65];   // logits fp32
  __shared__ __align__(16) _Float16 sP[64][72];   // softmax probs f16
  __shared__ __align__(16) _Float16 sV[128][72];  // v tile transposed [l][e]

  const int tid  = threadIdx.x;
  const int wid  = tid >> 5;
  const int lane = tid & 31;
  const int bh   = blockIdx.x;
  const int b    = bh >> 3;
  const int head = bh & 7;
  const size_t hb = ((size_t)b * CC + head * DHEAD) * LL;
  const _Float16* qh = q + hb;
  const _Float16* kh = k + hb;
  const _Float16* vh = v + hb;
  _Float16* oh       = o + hb;

  // ---- phase 1: logits. 16 tiles, 2 per wave (shared A fragment). ----
  const int mt  = wid & 3;
  const int nt0 = (wid >> 2) * 2;
  v8f s0 = zero8(), s1 = zero8();
  for (int kb = 0; kb < LL; kb += 32) {
    v16h af = frag_ld(qh, mt * 16, LL, kb);
    v16h b0 = frag_ld(kh, nt0 * 16, LL, kb);
    v16h b1 = frag_ld(kh, (nt0 + 1) * 16, LL, kb);
    s0 = wmma16(af, b0, s0);
    s1 = wmma16(af, b1, s1);
  }
  const float scale = 0.015625f;  // 1/sqrt(4096)
  #pragma unroll
  for (int r = 0; r < 8; ++r) {
    const int row = mt * 16 + r + ((lane >> 4) << 3);
    sS[row][nt0 * 16 + (lane & 15)]       = s0[r] * scale;
    sS[row][(nt0 + 1) * 16 + (lane & 15)] = s1[r] * scale;
  }
  __syncthreads();

  // ---- softmax over e (64 wide), one thread per row ----
  if (tid < 64) {
    float mx = -3.0e38f;
    for (int e = 0; e < 64; ++e) mx = fmaxf(mx, sS[tid][e]);
    float sum = 0.f;
    for (int e = 0; e < 64; ++e) {
      float t = __expf(sS[tid][e] - mx);
      sum += t;
      sS[tid][e] = t;
    }
    const float inv = 1.f / sum;
    for (int e = 0; e < 64; ++e) sP[tid][e] = (_Float16)(sS[tid][e] * inv);
  }
  __syncthreads();

  // ---- phase 2: O = P x v. Preload all P fragments (held in VGPRs). ----
  v16h pf[4][2];
  #pragma unroll
  for (int m2 = 0; m2 < 4; ++m2)
    #pragma unroll
    for (int ks = 0; ks < 2; ++ks)
      pf[m2][ks] = frag_ld(&sP[0][0], m2 * 16, 72, ks * 32);

  for (int l0 = 0; l0 < LL; l0 += 128) {
    __syncthreads();
    // stage v[e=0..63][l0..+127] transposed into sV[l][e]
    {
      const int e = tid >> 2, seg = tid & 3;
      const _Float16* src = vh + (size_t)e * LL + l0 + seg * 32;
      #pragma unroll
      for (int q4 = 0; q4 < 4; ++q4) {
        v8h x = *(const v8h*)(src + q4 * 8);
        #pragma unroll
        for (int j = 0; j < 8; ++j) sV[seg * 32 + q4 * 8 + j][e] = x[j];
      }
    }
    __syncthreads();

    // each wave owns a 16-wide l sub-chunk: n_local = wid*16 + lane%16
    v8f oacc[4];
    #pragma unroll
    for (int m2 = 0; m2 < 4; ++m2) oacc[m2] = zero8();
    #pragma unroll
    for (int ks = 0; ks < 2; ++ks) {
      v16h bf = frag_ld(&sV[0][0], wid * 16, 72, ks * 32);
      #pragma unroll
      for (int m2 = 0; m2 < 4; ++m2)
        oacc[m2] = wmma16(pf[m2][ks], bf, oacc[m2]);
    }
    #pragma unroll
    for (int m2 = 0; m2 < 4; ++m2)
      #pragma unroll
      for (int r = 0; r < 8; ++r) {
        const int c = m2 * 16 + r + ((lane >> 4) << 3);
        oh[(size_t)c * LL + l0 + wid * 16 + (lane & 15)] = (_Float16)oacc[m2][r];
      }
  }
}

// ---------------------------------------------------------------------------
extern "C" void kernel_launch(void* const* d_in, const int* in_sizes, int n_in,
                              void* d_out, int out_size, void* d_ws, size_t ws_size,
                              hipStream_t stream) {
  const float* query = (const float*)d_in[0];
  const float* key_  = (const float*)d_in[1];
  const float* value = (const float*)d_in[2];
  const float* wq = (const float*)d_in[3];
  const float* bq = (const float*)d_in[4];
  const float* wk = (const float*)d_in[5];
  const float* bk = (const float*)d_in[6];
  const float* wv = (const float*)d_in[7];
  const float* bv = (const float*)d_in[8];
  const float* wo = (const float*)d_in[9];
  const float* bo = (const float*)d_in[10];
  float* out = (float*)d_out;

  // f16 intermediates in workspace: q,k,v,attn-out (B*C*L halfs = 32 MiB each)
  const size_t S = (size_t)BB * CC * LL;
  _Float16* q16 = (_Float16*)d_ws;
  _Float16* k16 = q16 + S;
  _Float16* v16 = k16 + S;
  _Float16* o16 = v16 + S;

  dim3 blk(256);
  dim3 grd(LL / 128, CC / 128, BB);  // (32, 4, 8)

  // The three projections use three DIFFERENT inputs (wq.query, wk.key,
  // wv.value), so nothing is shared between them: one GEMM per launch.
  panel_gemm<false, false><<<grd, blk, 0, stream>>>(wq, bq, query, nullptr,
                                                    nullptr, q16);
  panel_gemm<false, false><<<grd, blk, 0, stream>>>(wk, bk, key_, nullptr,
                                                    nullptr, k16);
  panel_gemm<false, false><<<grd, blk, 0, stream>>>(wv, bv, value, nullptr,
                                                    nullptr, v16);

  attn64<<<dim3(BB * NHEAD), blk, 0, stream>>>(q16, k16, v16, o16);

  panel_gemm<true, true><<<grd, blk, 0, stream>>>(wo, bo, nullptr, o16,
                                                  out, nullptr);
}
